// MambaBlock_23003844837818
// MI455X (gfx1250) — compile-verified, matrix operand order checked by reading, not run on hardware
//
#include <hip/hip_runtime.h>
#include <hip/hip_bf16.h>
#include <math.h>

typedef __bf16 bf16;
typedef __attribute__((ext_vector_type(16))) __bf16 v16bf;
typedef __attribute__((ext_vector_type(8)))  float  v8f;

#define L_SEQ 2048
#define D_M   768
#define E_D   1536
#define N_XZ  3072      // 2*ED
#define DT_R  48
#define DT_RP 64        // DT_RANK padded to K-multiple of 32
#define D_ST  16
#define F_DBC 80        // DT_RANK + 2*D_STATE
#define NCHUNK 32
#define CHUNK  64       // L_SEQ / NCHUNK

__device__ __forceinline__ float silu_f(float v) { return v / (1.0f + __expf(-v)); }
__device__ __forceinline__ float softplus_f(float v) {
  return (v > 20.0f) ? v : log1pf(__expf(v));
}

// ---- gfx1250 async copy: global -> LDS via ASYNCcnt DMA path ----
__device__ __forceinline__ void async_copy_b128(void* lds_dst, const void* gsrc) {
  uint32_t lds_off = (uint32_t)(uintptr_t)lds_dst;   // low 32 bits of generic LDS ptr = LDS byte addr
  asm volatile("global_load_async_to_lds_b128 %0, %1, off"
               :: "v"(lds_off), "v"(gsrc) : "memory");
}
__device__ __forceinline__ void wait_async0() {
  asm volatile("s_wait_asynccnt 0" ::: "memory");
}

// ---------------- elementwise helpers ----------------

__global__ void k_cast_bf16(const float* __restrict__ s, bf16* __restrict__ d, int n) {
  int i = blockIdx.x * blockDim.x + threadIdx.x;
  if (i < n) d[i] = (bf16)s[i];
}

__global__ void k_cast_pad(const float* __restrict__ s, int lds,
                           bf16* __restrict__ d, int ldd, int rows, int cols) {
  int i = blockIdx.x * blockDim.x + threadIdx.x;
  int r = i / ldd, c = i % ldd;
  if (r < rows) d[i] = (c < cols) ? (bf16)s[(size_t)r * lds + c] : (bf16)0.0f;
}

// depthwise causal conv (D_CONV=4) + bias + SiLU; reads xb = xz[:, :ED]
__global__ void k_conv_silu(const float* __restrict__ xz,
                            const float* __restrict__ conv_w,
                            const float* __restrict__ conv_b,
                            float* __restrict__ xc_f, bf16* __restrict__ xc_b) {
  int i = blockIdx.x * blockDim.x + threadIdx.x;
  if (i >= L_SEQ * E_D) return;
  int l = i / E_D, e = i % E_D;
  float acc = conv_b[e];
#pragma unroll
  for (int k = 0; k < 4; ++k) {
    int ls = l - 3 + k;
    if (ls >= 0) acc += conv_w[e * 4 + k] * xz[(size_t)ls * N_XZ + e];
  }
  float v = silu_f(acc);
  xc_f[i] = v;
  xc_b[i] = (bf16)v;
}

// gate: yz = bf16( y * silu(z) ), z = xz[:, ED:]
__global__ void k_gate(const float* __restrict__ y, const float* __restrict__ xz,
                       bf16* __restrict__ yz) {
  int i = blockIdx.x * blockDim.x + threadIdx.x;
  if (i >= L_SEQ * E_D) return;
  int l = i / E_D, e = i % E_D;
  float z = xz[(size_t)l * N_XZ + E_D + e];
  yz[i] = (bf16)(y[i] * silu_f(z));
}

// ---------------- WMMA GEMM (small-N fallback): C[M,N] = A[M,K]*B[N,K]^T ----
// One wave per 16x16 tile, fragments straight from global. Used for N=80.
__global__ void k_wmma_gemm_nt16(const bf16* __restrict__ A, const bf16* __restrict__ B,
                                 float* __restrict__ C, int M, int N, int K) {
  int lane = threadIdx.x & 31;
  int wave = threadIdx.x >> 5;
  int tilesN = N >> 4;
  int tile = blockIdx.x * (blockDim.x >> 5) + wave;
  if (tile >= (M >> 4) * tilesN) return;              // wave-uniform: EXEC stays all-1s
  int tm = tile / tilesN, tn = tile % tilesN;
  int row = lane & 15, half = lane >> 4;

  const bf16* pa = A + (size_t)(tm * 16 + row) * K + half * 8;
  const bf16* pb = B + (size_t)(tn * 16 + row) * K + half * 8;

  v8f acc = {};
  for (int k = 0; k < K; k += 32) {
    __builtin_prefetch(pa + k + 128, 0, 0);
    __builtin_prefetch(pb + k + 128, 0, 0);
    v16bf a, b;
#pragma unroll
    for (int i = 0; i < 8; ++i) { a[i] = pa[k + i]; a[8 + i] = pa[k + 16 + i]; }
#pragma unroll
    for (int i = 0; i < 8; ++i) { b[i] = pb[k + i]; b[8 + i] = pb[k + 16 + i]; }
    acc = __builtin_amdgcn_wmma_f32_16x16x32_bf16(false, a, false, b,
                                                  (short)0, acc, false, false);
  }
  float* pc = C + (size_t)(tm * 16 + half * 8) * N + tn * 16 + row;
#pragma unroll
  for (int r = 0; r < 8; ++r) pc[(size_t)r * N] = acc[r];
}

// ---------------- WMMA GEMM with async-LDS A staging ----------------
// Block = 128 threads = 4 waves. Block tile: 16(M) x 256(N); wave tile 16x64.
// The 16x32 A tile per K-step is DMA'd to LDS once (double-buffered) and
// shared by all 4 waves; each wave reuses its A fragment across 4 WMMAs.
__global__ void k_wmma_gemm_lds(const bf16* __restrict__ A, const bf16* __restrict__ B,
                                float* __restrict__ C, int M, int N, int K) {
  __shared__ bf16 sA[2][16 * 32];                     // 2 x 1KB double buffer
  const int tid  = threadIdx.x;
  const int lane = tid & 31;
  const int wave = tid >> 5;
  const int nBlk = N >> 8;                            // 256-wide N block tiles
  const int tm   = blockIdx.x / nBlk;                 // 16-row M tile
  const int bn   = (blockIdx.x % nBlk) * 256 + wave * 64;
  const int row  = lane & 15, half = lane >> 4;

  // staging lanes: tid<64, each moves 16B: row = tid>>2, 8 bf16 at col (tid&3)*8
  const int srow = tid >> 2, scol = (tid & 3) * 8;
  const bf16* gA = A + (size_t)(tm * 16 + srow) * K + scol;

  if (tid < 64) async_copy_b128(&sA[0][tid * 8], gA);
  wait_async0();
  __syncthreads();

  const bf16* pb = B + (size_t)(bn + row) * K + half * 8;
  v8f acc[4] = {{}, {}, {}, {}};

  for (int k = 0; k < K; k += 32) {
    int cur = (k >> 5) & 1;
    if (k + 32 < K && tid < 64)
      async_copy_b128(&sA[cur ^ 1][tid * 8], gA + k + 32);

    // A fragment from LDS (shared by the 4 waves)
    v16bf a;
    const bf16* la = &sA[cur][row * 32 + half * 8];
#pragma unroll
    for (int i = 0; i < 8; ++i) { a[i] = la[i]; a[8 + i] = la[16 + i]; }

    // 4 B fragments from global, 4 WMMAs reusing a
#pragma unroll
    for (int j = 0; j < 4; ++j) {
      const bf16* pbj = pb + (size_t)j * 16 * K + k;
      __builtin_prefetch(pbj + 64, 0, 0);
      v16bf b;
#pragma unroll
      for (int i = 0; i < 8; ++i) { b[i] = pbj[i]; b[8 + i] = pbj[16 + i]; }
      acc[j] = __builtin_amdgcn_wmma_f32_16x16x32_bf16(false, a, false, b,
                                                       (short)0, acc[j], false, false);
    }

    wait_async0();       // next buffer landed; cur fully consumed by this wave
    __syncthreads();     // all waves done with cur before it is overwritten
  }

#pragma unroll
  for (int j = 0; j < 4; ++j) {
    float* pc = C + (size_t)(tm * 16 + half * 8) * N + bn + j * 16 + row;
#pragma unroll
    for (int r = 0; r < 8; ++r) pc[(size_t)r * N] = acc[j][r];
  }
}

// ---------------- chunked selective scan (3 passes) ----------------
// h[l] = a[l]*h[l-1] + b[l];  a = exp(softplus(dd+dtb)*A), b = delta*B*xc.
// Pass A: per (chunk,e,n) summaries (prodA, scanB).  32x parallelism.
__global__ void k_scan_chunk(const float* __restrict__ ddl, const float* __restrict__ dt_b,
                             const float* __restrict__ dbc, const float* __restrict__ xc,
                             const float* __restrict__ A_log,
                             float* __restrict__ Ac, float* __restrict__ Bc) {
  int t = blockIdx.x * blockDim.x + threadIdx.x;      // [0, NCHUNK*ED*16)
  int c  = t / (E_D * D_ST);
  int en = t % (E_D * D_ST);
  int e = en >> 4, n = en & 15;
  float Aen = -__expf(A_log[e * D_ST + n]);
  float dtb = dt_b[e];
  float Ap = 1.0f, Bp = 0.0f;
  int l0 = c * CHUNK;
  for (int i = 0; i < CHUNK; ++i) {
    int l = l0 + i;
    float delta = softplus_f(ddl[(size_t)l * E_D + e] + dtb);
    float a = __expf(delta * Aen);
    float b = delta * dbc[(size_t)l * F_DBC + DT_R + n] * xc[(size_t)l * E_D + e];
    Ap *= a;
    Bp = a * Bp + b;
  }
  Ac[t] = Ap;
  Bc[t] = Bp;
}

// Pass B: tiny 32-step cross-chunk scan per channel; stores chunk-entry state.
__global__ void k_scan_fix(const float* __restrict__ Ac, const float* __restrict__ Bc,
                           float* __restrict__ h0) {
  int en = blockIdx.x * blockDim.x + threadIdx.x;     // [0, ED*16)
  float h = 0.0f;
  for (int c = 0; c < NCHUNK; ++c) {
    size_t idx = (size_t)c * (E_D * D_ST) + en;
    h0[idx] = h;
    h = Ac[idx] * h + Bc[idx];
  }
}

// Pass C: replay each chunk from its entry state; reduce over n (16 lanes) and
// emit y = sum_n h*C + Dp*xc.  Lanes 0..15 / 16..31 of a wave share (chunk,e).
__global__ void k_scan_apply(const float* __restrict__ ddl, const float* __restrict__ dt_b,
                             const float* __restrict__ dbc, const float* __restrict__ xc,
                             const float* __restrict__ A_log, const float* __restrict__ Dp,
                             const float* __restrict__ h0, float* __restrict__ y) {
  int t = blockIdx.x * blockDim.x + threadIdx.x;
  int c  = t / (E_D * D_ST);
  int en = t % (E_D * D_ST);
  int e = en >> 4, n = en & 15;
  float Aen = -__expf(A_log[e * D_ST + n]);
  float dtb = dt_b[e], dpe = Dp[e];
  float h = h0[(size_t)c * (E_D * D_ST) + en];
  int l0 = c * CHUNK;
  for (int i = 0; i < CHUNK; ++i) {
    int l = l0 + i;
    float delta = softplus_f(ddl[(size_t)l * E_D + e] + dtb);
    float xcv = xc[(size_t)l * E_D + e];
    h = __expf(delta * Aen) * h + delta * dbc[(size_t)l * F_DBC + DT_R + n] * xcv;
    float contrib = h * dbc[(size_t)l * F_DBC + DT_R + D_ST + n];
#pragma unroll
    for (int off = 8; off >= 1; off >>= 1)
      contrib += __shfl_xor(contrib, off, 32);        // stays within 16-lane group
    if (n == 0) y[(size_t)l * E_D + e] = contrib + dpe * xcv;
  }
}

// ---------------- launcher ----------------

extern "C" void kernel_launch(void* const* d_in, const int* in_sizes, int n_in,
                              void* d_out, int out_size, void* d_ws, size_t ws_size,
                              hipStream_t stream) {
  const float* x      = (const float*)d_in[0];
  const float* W_in   = (const float*)d_in[1];
  const float* conv_w = (const float*)d_in[2];
  const float* conv_b = (const float*)d_in[3];
  const float* W_x    = (const float*)d_in[4];
  const float* dt_w   = (const float*)d_in[5];
  const float* dt_b   = (const float*)d_in[6];
  const float* A_log  = (const float*)d_in[7];
  const float* Dp     = (const float*)d_in[8];
  const float* W_out  = (const float*)d_in[9];
  float* out = (float*)d_out;

  char*  ws  = (char*)d_ws;
  size_t off = 0;
  auto carve = [&](size_t bytes) { void* p = ws + off; off = (off + bytes + 255) & ~(size_t)255; return p; };

  float* xz      = (float*)carve((size_t)L_SEQ * N_XZ * 4);
  bf16*  x_bf    = (bf16*) carve((size_t)L_SEQ * D_M * 2);
  bf16*  Win_bf  = (bf16*) carve((size_t)N_XZ * D_M * 2);
  bf16*  Wx_bf   = (bf16*) carve((size_t)F_DBC * E_D * 2);
  bf16*  Wout_bf = (bf16*) carve((size_t)D_M * E_D * 2);
  bf16*  dtw_pad = (bf16*) carve((size_t)E_D * DT_RP * 2);
  float* xc_f    = (float*)carve((size_t)L_SEQ * E_D * 4);
  bf16*  xc_b    = (bf16*) carve((size_t)L_SEQ * E_D * 2);
  float* dbc     = (float*)carve((size_t)L_SEQ * F_DBC * 4);
  bf16*  dpad    = (bf16*) carve((size_t)L_SEQ * DT_RP * 2);
  float* ddl     = (float*)carve((size_t)L_SEQ * E_D * 4);
  float* y       = (float*)carve((size_t)L_SEQ * E_D * 4);
  bf16*  yz_b    = (bf16*) carve((size_t)L_SEQ * E_D * 2);
  float* Ac      = (float*)carve((size_t)NCHUNK * E_D * D_ST * 4);
  float* Bc      = (float*)carve((size_t)NCHUNK * E_D * D_ST * 4);
  float* h0      = (float*)carve((size_t)NCHUNK * E_D * D_ST * 4);

  const int T = 256;
  auto blocks = [](int n, int t) { return (n + t - 1) / t; };

  // bf16 staging
  k_cast_bf16<<<blocks(L_SEQ * D_M, T), T, 0, stream>>>(x, x_bf, L_SEQ * D_M);
  k_cast_bf16<<<blocks(N_XZ * D_M, T), T, 0, stream>>>(W_in, Win_bf, N_XZ * D_M);
  k_cast_bf16<<<blocks(F_DBC * E_D, T), T, 0, stream>>>(W_x, Wx_bf, F_DBC * E_D);
  k_cast_bf16<<<blocks(D_M * E_D, T), T, 0, stream>>>(W_out, Wout_bf, D_M * E_D);
  k_cast_pad<<<blocks(E_D * DT_RP, T), T, 0, stream>>>(dt_w, DT_R, dtw_pad, DT_RP, E_D, DT_R);

  // GEMM1: xz(L,3072) = x(L,768) * W_in^T   (async-LDS A staging)
  k_wmma_gemm_lds<<<(L_SEQ / 16) * (N_XZ / 256), 128, 0, stream>>>(x_bf, Win_bf, xz, L_SEQ, N_XZ, D_M);

  // conv + SiLU
  k_conv_silu<<<blocks(L_SEQ * E_D, T), T, 0, stream>>>(xz, conv_w, conv_b, xc_f, xc_b);

  // GEMM2: dbc(L,80) = xc * W_x^T  (N=80 -> 16-wide fallback; 640 tiles)
  k_wmma_gemm_nt16<<<(128 * (F_DBC / 16)) / 8, T, 0, stream>>>(xc_b, Wx_bf, dbc, L_SEQ, F_DBC, E_D);

  // delta staging: dbc[:, :48] -> K=64 zero-padded bf16
  k_cast_pad<<<blocks(L_SEQ * DT_RP, T), T, 0, stream>>>(dbc, F_DBC, dpad, DT_RP, L_SEQ, DT_R);

  // GEMM3: ddl(L,1536) = delta_pad * dt_w_pad^T
  k_wmma_gemm_lds<<<(L_SEQ / 16) * (E_D / 256), 128, 0, stream>>>(dpad, dtw_pad, ddl, L_SEQ, E_D, DT_RP);

  // chunked scan: summaries -> cross-chunk fixup -> replay+reduce
  k_scan_chunk<<<blocks(NCHUNK * E_D * D_ST, T), T, 0, stream>>>(ddl, dt_b, dbc, xc_f, A_log, Ac, Bc);
  k_scan_fix<<<blocks(E_D * D_ST, T), T, 0, stream>>>(Ac, Bc, h0);
  k_scan_apply<<<blocks(NCHUNK * E_D * D_ST, T), T, 0, stream>>>(ddl, dt_b, dbc, xc_f, A_log, Dp, h0, y);

  // gate + cast
  k_gate<<<blocks(L_SEQ * E_D, T), T, 0, stream>>>(y, xz, yz_b);

  // GEMM4: out(L,768) = yz * W_out^T
  k_wmma_gemm_lds<<<(L_SEQ / 16) * (D_M / 256), 128, 0, stream>>>(yz_b, Wout_bf, out, L_SEQ, D_M, E_D);
}
